// DMR_47004122087444
// MI455X (gfx1250) — compile-verified, matrix operand order checked by reading, not run on hardware
//
#include <hip/hip_runtime.h>
#include <hip/hip_bf16.h>

#define B_   512
#define T_   200
#define TP   224      // T padded to 14*16 (= 7*32 for WMMA K loops)
#define E_   128
#define H1   80
#define H2   40
#define K1   512      // 4*E feats width
#define NEGV (-4294967295.0f)   // -2^32 + 1

typedef __bf16 bf16;
typedef __attribute__((ext_vector_type(16))) __bf16 v16bf;
typedef __attribute__((ext_vector_type(8)))  float  v8f;

// --- per-lane K index for 16-bit WMMA A/B operands (CDNA5 ISA 7.12.2) ---
// pair p (vector elems 2p,2p+1): k = (p<4 ? 2p : 16+2(p-4)) + 8*half
__device__ __forceinline__ int wmma_k(int p, int half) {
    return ((p < 4) ? 2 * p : 16 + 2 * (p - 4)) + 8 * half;
}

// ---------------------------------------------------------------------------
// Gather item_table rows directly into pre-swizzled B-operand layout:
// hisS[((((b*(E/16) + nt)*(TP/32) + kt)*32 + lane)*16 + idx]
// so k_dm_att's B tiles are single contiguous 32-byte loads.
// ---------------------------------------------------------------------------
__global__ void k_gatherS(const int* __restrict__ item_his,
                          const float* __restrict__ table,
                          bf16* __restrict__ hisS) {
    size_t gid = (size_t)blockIdx.x * 256 + threadIdx.x;
    size_t total = (size_t)B_ * TP * E_;
    if (gid >= total) return;
    int idx  = (int)(gid & 15);
    int lane = (int)((gid >> 4) & 31);
    size_t r = gid >> 9;
    int kt = (int)(r % (TP / 32));
    size_t r2 = r / (TP / 32);
    int nt = (int)(r2 % (E_ / 16));
    int b  = (int)(r2 / (E_ / 16));
    int p = idx >> 1, half = lane >> 4, col = lane & 15;
    int t = kt * 32 + wmma_k(p, half) + (idx & 1);
    int e = nt * 16 + col;
    float v = 0.f;
    if (t < T_) {
        int it = item_his[(size_t)b * T_ + t];
        v = table[(size_t)it * E_ + e];
    }
    hisS[gid] = (bf16)v;
}

// ---------------------------------------------------------------------------
// Small (rows,128)@(128,128) row GEMM, optional bias+prelu. grid=TP, block=128
// ---------------------------------------------------------------------------
__global__ void k_rowgemm(const float* __restrict__ A, const float* __restrict__ W,
                          const float* __restrict__ bias, const float* __restrict__ slope,
                          float* __restrict__ out, int rows, int applyAct) {
    int t = blockIdx.x, e = threadIdx.x;
    float s = 0.f;
    if (t < rows) {
        for (int p = 0; p < E_; ++p) s += A[t * E_ + p] * W[p * E_ + e];
        if (applyAct) {
            s += bias[e];
            float a = slope[0];
            s = (s >= 0.f) ? s : a * s;
        }
    }
    out[t * E_ + e] = s;
}

// itemW[b][e] = item_table[item[b]] @ fa_qW[:E]   grid=B_, block=128
__global__ void k_itemW(const int* __restrict__ item, const float* __restrict__ table,
                        const float* __restrict__ W, float* __restrict__ out) {
    int b = blockIdx.x, e = threadIdx.x;
    int it = item[b];
    float s = 0.f;
    for (int k = 0; k < E_; ++k) s += table[(size_t)it * E_ + k] * W[k * E_ + e];
    out[b * E_ + e] = s;
}

// ---------------------------------------------------------------------------
// Pre-swizzle a (K x N) fp32 weight into bf16 B-operand tiles:
// out[((nt*(K/32)+kt)*32 + lane)*16 + idx]  ->  one contiguous v16bf per lane
// ---------------------------------------------------------------------------
__global__ void k_swizzleB(const float* __restrict__ W, bf16* __restrict__ out,
                           int K, int N) {
    int gid = blockIdx.x * 256 + threadIdx.x;
    if (gid >= K * N) return;
    int idx  = gid & 15;
    int lane = (gid >> 4) & 31;
    int kt   = (gid >> 9) % (K / 32);
    int nt   = (gid >> 9) / (K / 32);
    int p = idx >> 1, half = lane >> 4, col = lane & 15;
    int kk = kt * 32 + wmma_k(p, half) + (idx & 1);
    int n  = nt * 16 + col;
    out[gid] = (bf16)W[kk * N + n];
}

// ---------------------------------------------------------------------------
// Fused att-MLP: 16 (b,t) rows per block. feats=[q,his,q-his,q*his] (512)
// L1: WMMA bf16 (16x512)@(512,80) -> sigmoid ; L2,L3: VALU.
// 5 waves (160 threads), one 16-col N-tile each.
// ---------------------------------------------------------------------------
__global__ __launch_bounds__(160)
void k_attmlp(const int* __restrict__ item_his, const float* __restrict__ table,
              const float* __restrict__ qdm,
              const float* __restrict__ posW, const float* __restrict__ itemW,
              const float* __restrict__ qb, const float* __restrict__ qa, int faMode,
              const bf16* __restrict__ W1s, const float* __restrict__ b1,
              const float* __restrict__ W2, const float* __restrict__ b2,
              const float* __restrict__ W3, const float* __restrict__ b3,
              float* __restrict__ sOut) {
    __shared__ bf16  feats[16][K1 + 8];   // +8 pad: bank-conflict-free column reads
    __shared__ float h1[16][H1];
    __shared__ float h2[16][H2];

    int b  = blockIdx.y;
    int t0 = blockIdx.x * 16;
    int tid = threadIdx.x;

    for (int idx = tid; idx < 16 * E_; idx += blockDim.x) {
        int r = idx / E_, e = idx % E_;
        int t = t0 + r;                       // t <= 207 < TP
        float qv;
        if (faMode) {
            float x = itemW[b * E_ + e] + posW[t * E_ + e] + qb[e];
            qv = (x >= 0.f) ? x : qa[0] * x;
        } else {
            qv = qdm[t * E_ + e];
        }
        float hv = 0.f;
        if (t < T_) {
            int it = item_his[(size_t)b * T_ + t];
            hv = table[(size_t)it * E_ + e];   // L2-resident gather, e-coalesced
        }
        feats[r][e]          = (bf16)qv;
        feats[r][E_ + e]     = (bf16)hv;
        feats[r][2 * E_ + e] = (bf16)(qv - hv);
        feats[r][3 * E_ + e] = (bf16)(qv * hv);
    }
    __syncthreads();

    int wave = tid >> 5, lane = tid & 31;
    int col = lane & 15, half = lane >> 4;
    int n0 = wave * 16;

    v8f c = {0.f, 0.f, 0.f, 0.f, 0.f, 0.f, 0.f, 0.f};
    for (int kt = 0; kt < K1 / 32; ++kt) {
        v16bf a;
#pragma unroll
        for (int p = 0; p < 8; ++p) {
            int kk = kt * 32 + wmma_k(p, half);
            a[2 * p]     = feats[col][kk];
            a[2 * p + 1] = feats[col][kk + 1];
        }
        v16bf bm = *reinterpret_cast<const v16bf*>(
            W1s + (((size_t)(wave * (K1 / 32) + kt)) * 32 + lane) * 16);
        c = __builtin_amdgcn_wmma_f32_16x16x32_bf16(false, a, false, bm,
                                                    (short)0, c, false, false);
    }
#pragma unroll
    for (int r = 0; r < 8; ++r) {
        int row = r + 8 * half;
        float x = c[r] + b1[n0 + col];
        h1[row][n0 + col] = 1.0f / (1.0f + __expf(-x));
    }
    __syncthreads();

    for (int idx = tid; idx < 16 * H2; idx += blockDim.x) {
        int r = idx / H2, j = idx % H2;
        float s = b2[j];
        for (int k = 0; k < H1; ++k) s += h1[r][k] * W2[k * H2 + j];
        h2[r][j] = 1.0f / (1.0f + __expf(-s));
    }
    __syncthreads();

    if (tid < 16) {
        float s = b3[0];
        for (int j = 0; j < H2; ++j) s += h2[tid][j] * W3[j];
        int t = t0 + tid;
        if (t < TP) sOut[(size_t)b * TP + t] = s;
    }
}

// ---------------------------------------------------------------------------
// Per-b: dm_scores (masked), tril-softmax st, att=st@his (WMMA, swizzled his),
// out=prelu(att@oW) (WMMA), masked row-sum -> dm_user_vector
// 8 waves = 8 N-tiles over E=128.
// ---------------------------------------------------------------------------
__global__ __launch_bounds__(256)
void k_dm_att(const float* __restrict__ sdm, const unsigned char* __restrict__ mask,
              const bf16* __restrict__ hisS, const bf16* __restrict__ oWs,
              const float* __restrict__ ob, const float* __restrict__ oa,
              float* __restrict__ outUser, float* __restrict__ outScores) {
    __shared__ float sv[TP];
    __shared__ float Mi[T_];
    __shared__ float Si[T_];
    __shared__ bf16  stT[16][TP + 8];
    __shared__ bf16  attT[16][E_ + 8];
    __shared__ float acc[E_];

    int b = blockIdx.x;
    int tid = threadIdx.x;

    for (int t = tid; t < TP; t += 256) {
        float v = NEGV;
        if (t < T_) {
            float s = sdm[(size_t)b * TP + t];
            v = mask[(size_t)b * T_ + t] ? s : NEGV;
            outScores[(size_t)b * T_ + t] = v;          // dm_scores output
        }
        sv[t] = v;
    }
    if (tid < E_) acc[tid] = 0.f;
    __syncthreads();

    if (tid == 0) {   // serial prefix (max, sum-exp) scan, T=200
        float m = -3.0e38f, s = 0.f;
        for (int j = 0; j < T_; ++j) {
            float x = sv[j];
            float mn = fmaxf(m, x);
            s = s * __expf(m - mn) + __expf(x - mn);
            m = mn;
            Mi[j] = m; Si[j] = s;
        }
    }
    __syncthreads();

    int wave = tid >> 5, lane = tid & 31;
    int col = lane & 15, half = lane >> 4;
    int n0 = wave * 16;
    float aSlope = oa[0];

    for (int m = 0; m < 13; ++m) {                 // 13 tiles cover rows 0..207
        for (int idx = tid; idx < 16 * TP; idx += 256) {
            int r = idx / TP, j = idx % TP;
            int i = m * 16 + r;
            float v = 0.f;
            if (i < T_ && j <= i) v = __expf(sv[j] - Mi[i]) / Si[i];
            stT[r][j] = (bf16)v;
        }
        __syncthreads();

        // att tile = st(16xTP) @ his(TPxE), this wave's 16 columns
        v8f c = {0.f, 0.f, 0.f, 0.f, 0.f, 0.f, 0.f, 0.f};
        for (int kt = 0; kt < TP / 32; ++kt) {
            v16bf a;
#pragma unroll
            for (int p = 0; p < 8; ++p) {
                int kk = kt * 32 + wmma_k(p, half);
                a[2 * p]     = stT[col][kk];
                a[2 * p + 1] = stT[col][kk + 1];
            }
            v16bf bm = *reinterpret_cast<const v16bf*>(
                hisS + ((((size_t)b * (E_ / 16) + wave) * (TP / 32) + kt) * 32 + lane) * 16);
            c = __builtin_amdgcn_wmma_f32_16x16x32_bf16(false, a, false, bm,
                                                        (short)0, c, false, false);
        }
#pragma unroll
        for (int r = 0; r < 8; ++r) attT[r + 8 * half][n0 + col] = (bf16)c[r];
        __syncthreads();

        // out tile = att(16x128) @ oW(128x128), prelu, masked row-sum
        v8f d = {0.f, 0.f, 0.f, 0.f, 0.f, 0.f, 0.f, 0.f};
        for (int kt = 0; kt < E_ / 32; ++kt) {
            v16bf a;
#pragma unroll
            for (int p = 0; p < 8; ++p) {
                int kk = kt * 32 + wmma_k(p, half);
                a[2 * p]     = attT[col][kk];
                a[2 * p + 1] = attT[col][kk + 1];
            }
            v16bf bm = *reinterpret_cast<const v16bf*>(
                oWs + (((size_t)(wave * (E_ / 32) + kt)) * 32 + lane) * 16);
            d = __builtin_amdgcn_wmma_f32_16x16x32_bf16(false, a, false, bm,
                                                        (short)0, d, false, false);
        }
#pragma unroll
        for (int r = 0; r < 8; ++r) {
            int row = r + 8 * half;
            int t = m * 16 + row;
            if (t < T_) {
                float x = d[r] + ob[n0 + col];
                x = (x >= 0.f) ? x : aSlope * x;
                atomicAdd(&acc[n0 + col], x);       // ds_add_f32
            }
        }
        __syncthreads();
    }
    if (tid < E_) outUser[(size_t)b * E_ + tid] = acc[tid];
}

// ---------------------------------------------------------------------------
// Per-b: fa softmax over T, alphas/scores_unnorm outputs,
// att_outputs = alphas @ his (fp32 gather for accuracy; M=1 -> VALU)
// ---------------------------------------------------------------------------
__global__ __launch_bounds__(256)
void k_fa_final(const float* __restrict__ sfa, const unsigned char* __restrict__ mask,
                const int* __restrict__ item_his, const float* __restrict__ table,
                float* __restrict__ outAtt, float* __restrict__ outAlphas,
                float* __restrict__ outUnnorm) {
    __shared__ float sv2[T_];
    __shared__ float al[T_];
    __shared__ float red[256];
    int b = blockIdx.x, tid = threadIdx.x;

    float lmax = -3.0e38f;
    for (int t = tid; t < T_; t += 256) {
        float s = sfa[(size_t)b * TP + t];
        unsigned char mk = mask[(size_t)b * T_ + t];
        float v = mk ? s : NEGV;
        sv2[t] = v;
        outUnnorm[(size_t)b * T_ + t] = mk ? v : 0.f;  // where(km, s2m, 0)
        lmax = fmaxf(lmax, v);
    }
    red[tid] = lmax;
    __syncthreads();
    for (int off = 128; off > 0; off >>= 1) {
        if (tid < off) red[tid] = fmaxf(red[tid], red[tid + off]);
        __syncthreads();
    }
    float mx = red[0];
    __syncthreads();
    float lsum = 0.f;
    for (int t = tid; t < T_; t += 256) lsum += __expf(sv2[t] - mx);
    red[tid] = lsum;
    __syncthreads();
    for (int off = 128; off > 0; off >>= 1) {
        if (tid < off) red[tid] += red[tid + off];
        __syncthreads();
    }
    float inv = 1.0f / red[0];
    __syncthreads();
    for (int t = tid; t < T_; t += 256) {
        float a = __expf(sv2[t] - mx) * inv;
        al[t] = a;
        outAlphas[(size_t)b * T_ + t] = a;
    }
    __syncthreads();
    if (tid < E_) {
        float s = 0.f;
        for (int t = 0; t < T_; ++t) {
            int it = item_his[(size_t)b * T_ + t];
            s += al[t] * table[(size_t)it * E_ + tid];  // coalesced across e
        }
        outAtt[(size_t)b * E_ + tid] = s;
    }
}

// ---------------------------------------------------------------------------
extern "C" void kernel_launch(void* const* d_in, const int* in_sizes, int n_in,
                              void* d_out, int out_size, void* d_ws, size_t ws_size,
                              hipStream_t stream) {
    const int*   item       = (const int*)d_in[0];
    const int*   item_his   = (const int*)d_in[1];
    const unsigned char* mask = (const unsigned char*)d_in[2];
    const float* item_table = (const float*)d_in[3];
    const float* pos_table  = (const float*)d_in[4];
    const float* dm_pos     = (const float*)d_in[5];
    const float* dm_qW      = (const float*)d_in[6];
    const float* dm_qb      = (const float*)d_in[7];
    const float* dm_qa      = (const float*)d_in[8];
    const float* dm_W1      = (const float*)d_in[9];
    const float* dm_b1      = (const float*)d_in[10];
    const float* dm_W2      = (const float*)d_in[11];
    const float* dm_b2      = (const float*)d_in[12];
    const float* dm_W3      = (const float*)d_in[13];
    const float* dm_b3      = (const float*)d_in[14];
    const float* dm_oW      = (const float*)d_in[15];
    const float* dm_ob      = (const float*)d_in[16];
    const float* dm_oa      = (const float*)d_in[17];
    const float* fa_qW      = (const float*)d_in[18];
    const float* fa_qb      = (const float*)d_in[19];
    const float* fa_qa      = (const float*)d_in[20];
    const float* fa_W1      = (const float*)d_in[21];
    const float* fa_b1      = (const float*)d_in[22];
    const float* fa_W2      = (const float*)d_in[23];
    const float* fa_b2      = (const float*)d_in[24];
    const float* fa_W3      = (const float*)d_in[25];
    const float* fa_b3      = (const float*)d_in[26];

    // ---- workspace layout (256B aligned) ----
    char* ws = (char*)d_ws;
    size_t off = 0;
    auto take = [&](size_t bytes) {
        char* p = ws + off;
        off = (off + bytes + 255) & ~(size_t)255;
        return p;
    };
    bf16*  hisS  = (bf16*) take((size_t)B_ * TP * E_ * 2);   // swizzled B-operand
    float* qdm   = (float*)take((size_t)TP * E_ * 4);
    float* posW  = (float*)take((size_t)TP * E_ * 4);
    float* itemW = (float*)take((size_t)B_ * E_ * 4);
    bf16*  w1dm  = (bf16*) take((size_t)K1 * H1 * 2);
    bf16*  w1fa  = (bf16*) take((size_t)K1 * H1 * 2);
    bf16*  oWs   = (bf16*) take((size_t)E_ * E_ * 2);
    float* sdm   = (float*)take((size_t)B_ * TP * 4);
    float* sfa   = (float*)take((size_t)B_ * TP * 4);
    (void)ws_size;

    // ---- output layout (concatenated, return order) ----
    float* out      = (float*)d_out;
    float* o_user   = out;                              // (B,E)
    float* o_scores = out + (size_t)B_ * E_;            // (B,1,T)
    float* o_att    = o_scores + (size_t)B_ * T_;       // (B,E)
    float* o_alphas = o_att + (size_t)B_ * E_;          // (B,1,T)
    float* o_unnorm = o_alphas + (size_t)B_ * T_;       // (B,1,T)

    // 1) gather his embeddings directly into swizzled WMMA-B layout
    {
        size_t total = (size_t)B_ * TP * E_;
        k_gatherS<<<dim3((unsigned)((total + 255) / 256)), 256, 0, stream>>>(
            item_his, item_table, hisS);
    }
    // 2) batch-independent precomputes
    k_rowgemm<<<TP, 128, 0, stream>>>(dm_pos, dm_qW, dm_qb, dm_qa, qdm, T_, 1);
    k_rowgemm<<<TP, 128, 0, stream>>>(pos_table, fa_qW + E_ * E_, dm_qb, dm_qa,
                                      posW, T_, 0);
    k_itemW<<<B_, 128, 0, stream>>>(item, item_table, fa_qW, itemW);
    // 3) weight swizzles for WMMA B operands
    k_swizzleB<<<(K1 * H1 + 255) / 256, 256, 0, stream>>>(dm_W1, w1dm, K1, H1);
    k_swizzleB<<<(K1 * H1 + 255) / 256, 256, 0, stream>>>(fa_W1, w1fa, K1, H1);
    k_swizzleB<<<(E_ * E_ + 255) / 256, 256, 0, stream>>>(dm_oW, oWs, E_, E_);
    // 4) the two big fused MLPs (WMMA layer 1)
    k_attmlp<<<dim3(13, B_), 160, 0, stream>>>(item_his, item_table, qdm, posW, itemW,
                                               dm_qb, dm_qa, 0, w1dm, dm_b1, dm_W2,
                                               dm_b2, dm_W3, dm_b3, sdm);
    k_attmlp<<<dim3(13, B_), 160, 0, stream>>>(item_his, item_table, qdm, posW, itemW,
                                               fa_qb, fa_qa, 1, w1fa, fa_b1, fa_W2,
                                               fa_b2, fa_W3, fa_b3, sfa);
    // 5) dm tril-softmax attention + oW projection + user vector (WMMA x2)
    k_dm_att<<<B_, 256, 0, stream>>>(sdm, mask, hisS, oWs, dm_ob, dm_oa,
                                     o_user, o_scores);
    // 6) fa softmax + weighted-sum outputs
    k_fa_final<<<B_, 256, 0, stream>>>(sfa, mask, item_his, item_table,
                                       o_att, o_alphas, o_unnorm);
    (void)in_sizes; (void)n_in; (void)out_size;
}
